// sMGUCell_23545010717396
// MI455X (gfx1250) — matrix operand order repaired and checked by reading, not compiled, not run on hardware
//
#include <hip/hip_runtime.h>
#include <hip/hip_bf16.h>

// ---------------------------------------------------------------------------
// Fused sMGU/sLSTM cell for MI455X (gfx1250, wave32, WMMA bf16 16x16x32).
//
//   pre_f  = x@w_f + h@r_f + b_f      c~ = x@w_c + h@r_c + b_c
//   m_t = max(pre_f + m_prev, 0);  i' = exp(-m_t);  f' = exp(pre_f+m_prev-m_t)
//   c_t = f'*c_prev + i'*tanh(c~); n_t = f'*n_prev + i'; h_t = tanh(c_t/max(n_t,1e-8))
//
// One workgroup = 64 rows of B. The whole [x|h] activation panel (64x768)
// is converted f32->bf16 (v_perm-packed pairs -> ds_store_b32) and parked in
// LDS once (96 KB of the 320 KB/WGP), then all 8 column blocks of H are
// computed from it -> x/h read from HBM exactly once. Weight tiles stream
// through L2 into a double-buffered 2x8 KB LDS staging area (transposed
// [n][k]) so WMMA B fragments are contiguous ds_load_b128s, with one barrier
// per K-step and next-tile global loads overlapping the WMMAs.
// ---------------------------------------------------------------------------

typedef __attribute__((ext_vector_type(16))) __bf16 v16bf;
typedef __attribute__((ext_vector_type(8)))  __bf16 v8bf;
typedef __attribute__((ext_vector_type(8)))  float  v8f;

#define BDIM  65536
#define INDIM 256
#define HDIM  512
#define KTOT  (INDIM + HDIM)   // 768
#define BM    64
#define BN    64
#define BK    32
#define NSTEP (KTOT / BK)      // 24

// round-to-nearest-even f32 -> bf16, result in high 16 bits
__device__ __forceinline__ unsigned rne_hi(float f) {
  unsigned u = __float_as_uint(f);
  return u + (0x7FFFu + ((u >> 16) & 1u));
}
// pack two bf16 (lo, hi) into one dword with a single v_perm_b32
__device__ __forceinline__ unsigned pack_bf16(float lo, float hi) {
  return __builtin_amdgcn_perm(rne_hi(hi), rne_hi(lo), 0x07060302u);
}

__device__ __forceinline__ float fast_tanh(float x) {
  float e = __expf(2.0f * x);
  return (e - 1.0f) / (e + 1.0f);
}

__global__ __launch_bounds__(256)
void smgu_fused_kernel(const float* __restrict__ x,
                       const float* __restrict__ h_prev,
                       const float* __restrict__ c_prev,
                       const float* __restrict__ n_prev,
                       const float* __restrict__ m_prev,
                       const float* __restrict__ w_f,
                       const float* __restrict__ w_c,
                       const float* __restrict__ r_f,
                       const float* __restrict__ r_c,
                       const float* __restrict__ b_f,
                       const float* __restrict__ b_c,
                       float* __restrict__ out)
{
  // 64 x 768 bf16 activation panel (96 KB) + double-buffered B tiles (16 KB)
  __shared__ __align__(16) __bf16 lA[BM][KTOT];
  __shared__ __align__(16) __bf16 lB[2][2][BN][BK];   // [buf][gate][n][k]

  const int t = threadIdx.x;
  const int rowBase = blockIdx.x * BM;

  // ---- Stage activation panel: [x | h_prev] -> bf16 in LDS, read once ----
  // x part: 64 rows x 256 cols = 4096 float4
  for (int i = t; i < (BM * INDIM) / 4; i += 256) {
    const int r  = i >> 6;               // 64 float4 per row
    const int kc = (i & 63) << 2;
    const float4 v = *(const float4*)(x + (size_t)(rowBase + r) * INDIM + kc);
    unsigned* pA = (unsigned*)&lA[r][kc];
    pA[0] = pack_bf16(v.x, v.y);
    pA[1] = pack_bf16(v.z, v.w);
  }
  // h part: 64 rows x 512 cols = 8192 float4
  for (int i = t; i < (BM * HDIM) / 4; i += 256) {
    const int r  = i >> 7;               // 128 float4 per row
    const int kc = (i & 127) << 2;
    const float4 v = *(const float4*)(h_prev + (size_t)(rowBase + r) * HDIM + kc);
    unsigned* pA = (unsigned*)&lA[r][INDIM + kc];
    pA[0] = pack_bf16(v.x, v.y);
    pA[1] = pack_bf16(v.z, v.w);
  }

  // wave32 decomposition: 8 waves, each owns a 16x32 tile pair (f & c gates)
  const int lane = t & 31;
  const int half = lane >> 4;        // K-half / M-half selector
  const int l16  = lane & 15;
  const int w    = t >> 5;
  const int mi   = w & 3;            // 16-row subtile within 64
  const int njB  = (w >> 2) << 1;    // pair of 16-col subtiles within 64

  // B-tile staging assignment: thread -> (K row pair, 4-wide n chunk)
  const int kk2 = t >> 4;            // 0..15  (K pair index, k = 2*kk2)
  const int n0  = (t & 15) << 2;     // 0,4,..,60

  const size_t BH = (size_t)BDIM * HDIM;

  for (int cb = 0; cb < HDIM / BN; ++cb) {
    const int colBase = cb * BN;

    // Stage one 32x64 weight tile per gate into lB[buf], transposed [n][k],
    // bf16-pair packed so each store is a ds_store_b32. Branchless source
    // select between w_* (K<256) and r_* (K>=256).
    auto loadB = [&](int s, int buf) {
      const float* srcF = (s * BK < INDIM) ? w_f : r_f - (size_t)INDIM * HDIM;
      const float* srcC = (s * BK < INDIM) ? w_c : r_c - (size_t)INDIM * HDIM;
      const size_t rowk = (size_t)(s * BK + 2 * kk2);
      const float* pF = srcF + rowk * HDIM + colBase + n0;
      const float* pC = srcC + rowk * HDIM + colBase + n0;
      const float4 fa = *(const float4*)(pF);
      const float4 fb = *(const float4*)(pF + HDIM);
      const float4 ca = *(const float4*)(pC);
      const float4 cc = *(const float4*)(pC + HDIM);
      // prefetch the weight rows two K-steps ahead (L2 streaming)
      __builtin_prefetch(pF + 2 * BK * HDIM, 0, 1);
      __builtin_prefetch(pC + 2 * BK * HDIM, 0, 1);
      *(unsigned*)&lB[buf][0][n0 + 0][2 * kk2] = pack_bf16(fa.x, fb.x);
      *(unsigned*)&lB[buf][0][n0 + 1][2 * kk2] = pack_bf16(fa.y, fb.y);
      *(unsigned*)&lB[buf][0][n0 + 2][2 * kk2] = pack_bf16(fa.z, fb.z);
      *(unsigned*)&lB[buf][0][n0 + 3][2 * kk2] = pack_bf16(fa.w, fb.w);
      *(unsigned*)&lB[buf][1][n0 + 0][2 * kk2] = pack_bf16(ca.x, cc.x);
      *(unsigned*)&lB[buf][1][n0 + 1][2 * kk2] = pack_bf16(ca.y, cc.y);
      *(unsigned*)&lB[buf][1][n0 + 2][2 * kk2] = pack_bf16(ca.z, cc.z);
      *(unsigned*)&lB[buf][1][n0 + 3][2 * kk2] = pack_bf16(ca.w, cc.w);
    };

    v8f accF0 = {}, accF1 = {}, accC0 = {}, accC1 = {};

    loadB(0, 0);
    __syncthreads();   // also covers the lA panel staging on cb==0

    for (int s = 0; s < NSTEP; ++s) {
      const int cur = s & 1;
      if (s + 1 < NSTEP) loadB(s + 1, cur ^ 1);   // overlaps with WMMAs below

      // ---- A fragment (half selects K 0-15 vs 16-31, matching B)
      union { v16bf v; v8bf h[2]; } ua;
      const int arow = mi * 16 + l16;
      const int k0 = s * BK;
      ua.h[0] = *(const v8bf*)&lA[arow][k0 + half * 16];
      ua.h[1] = *(const v8bf*)&lA[arow][k0 + half * 16 + 8];

      // ---- B fragments (lane half selects K 0-15 vs 16-31, contiguous in LDS)
      const int kb = half << 4;
      const v16bf bF0 = *(const v16bf*)&lB[cur][0][(njB + 0) * 16 + l16][kb];
      const v16bf bF1 = *(const v16bf*)&lB[cur][0][(njB + 1) * 16 + l16][kb];
      const v16bf bC0 = *(const v16bf*)&lB[cur][1][(njB + 0) * 16 + l16][kb];
      const v16bf bC1 = *(const v16bf*)&lB[cur][1][(njB + 1) * 16 + l16][kb];

      accF0 = __builtin_amdgcn_wmma_f32_16x16x32_bf16(false, ua.v, false, bF0,
                                                      (short)0, accF0, false, false);
      accF1 = __builtin_amdgcn_wmma_f32_16x16x32_bf16(false, ua.v, false, bF1,
                                                      (short)0, accF1, false, false);
      accC0 = __builtin_amdgcn_wmma_f32_16x16x32_bf16(false, ua.v, false, bC0,
                                                      (short)0, accC0, false, false);
      accC1 = __builtin_amdgcn_wmma_f32_16x16x32_bf16(false, ua.v, false, bC1,
                                                      (short)0, accC1, false, false);
      __syncthreads();   // single barrier per K-step (double-buffered lB)
    }

    // ---- fused epilogue: C/D layout => VGPR r holds (M = r + 8*half, N = l16)
#pragma unroll
    for (int j = 0; j < 2; ++j) {
      const int hcol = colBase + (njB + j) * 16 + l16;
      const float bfv = b_f[hcol];
      const float bcv = b_c[hcol];
      const v8f accF = j ? accF1 : accF0;
      const v8f accC = j ? accC1 : accC0;
#pragma unroll
      for (int r = 0; r < 8; ++r) {
        const int brow = rowBase + mi * 16 + half * 8 + r;
        const size_t idx = (size_t)brow * HDIM + hcol;
        const float pre_f = accF[r] + bfv;
        const float c_til = accC[r] + bcv;
        const float mp  = m_prev[idx];
        const float m_t = fmaxf(pre_f + mp, 0.0f);
        const float i_p = __expf(-m_t);
        const float f_p = __expf(pre_f + mp - m_t);
        const float c_t = f_p * c_prev[idx] + i_p * fast_tanh(c_til);
        const float n_t = f_p * n_prev[idx] + i_p;
        const float h_t = fast_tanh(c_t / fmaxf(n_t, 1e-8f));
        out[idx]           = h_t;
        out[BH + idx]      = c_t;
        out[2 * BH + idx]  = n_t;
        out[3 * BH + idx]  = m_t;
      }
    }
  }
}

extern "C" void kernel_launch(void* const* d_in, const int* in_sizes, int n_in,
                              void* d_out, int out_size, void* d_ws, size_t ws_size,
                              hipStream_t stream) {
  (void)in_sizes; (void)n_in; (void)out_size; (void)d_ws; (void)ws_size;
  const float* x      = (const float*)d_in[0];
  const float* h_prev = (const float*)d_in[1];
  const float* c_prev = (const float*)d_in[2];
  const float* n_prev = (const float*)d_in[3];
  const float* m_prev = (const float*)d_in[4];
  const float* w_f    = (const float*)d_in[5];
  const float* w_c    = (const float*)d_in[6];
  const float* r_f    = (const float*)d_in[7];
  const float* r_c    = (const float*)d_in[8];
  const float* b_f    = (const float*)d_in[9];
  const float* b_c    = (const float*)d_in[10];
  float* out = (float*)d_out;

  dim3 grid(BDIM / BM);   // 1024 workgroups of 256 threads (8 wave32s)
  smgu_fused_kernel<<<grid, 256, 0, stream>>>(x, h_prev, c_prev, n_prev, m_prev,
                                              w_f, w_c, r_f, r_c, b_f, b_c, out);
}
